// DETR3D_Core_74062416052403
// MI455X (gfx1250) — compile-verified, matrix operand order checked by reading, not run on hardware
//
#include <hip/hip_runtime.h>
#include <hip/hip_bf16.h>
#include <cstdint>

#define BB 2
#define NN 6
#define CC 256
#define IMH 56
#define IMW 100
#define QQ 900
#define EE 256
#define NHH 8
#define HDD 32
#define RTOT (BB*QQ)      // 1800 rows
#define VT_LD 904         // padded row stride (halves) for transposed V, 16B aligned

typedef __attribute__((ext_vector_type(16))) _Float16 v16h;
typedef __attribute__((ext_vector_type(8)))  _Float16 v8h;
typedef __attribute__((ext_vector_type(8)))  float    v8f;

// Load a 16x32 f16 WMMA operand fragment from a row-major matrix.
// ISA 16-bit A layout: lane L -> row = (L&15); vector idx i<8 -> K = k0+8h+i,
// i>=8 -> K = k0+16+8h+(i-8), h = L>>4. Two contiguous 16B loads per lane.
// B operands use the symmetric layout with row = column index (so callers pass
// a transposed matrix).
__device__ __forceinline__ v16h frag_load(const _Float16* base, int row, int ld,
                                          int k0, int lane) {
  const int h = (lane >> 4) & 1;
  const _Float16* p = base + (size_t)row * ld + k0 + 8*h;
  v8h lo = *reinterpret_cast<const v8h*>(p);
  v8h hi = *reinterpret_cast<const v8h*>(p + 16);
  v16h r;
#pragma unroll
  for (int i = 0; i < 8; ++i) { r[i] = lo[i]; r[8+i] = hi[i]; }
  return r;
}

// ---- 1. Convert Wq/Wk/Wv/Wo (f32, ExE) to transposed f16 for B fragments ----
__global__ void k_prep_weights(const float* Wq, const float* Wk, const float* Wv,
                               const float* Wo, _Float16* Wt) {
  int idx = blockIdx.x*blockDim.x + threadIdx.x;
  if (idx >= 4*EE*EE) return;
  int w   = idx / (EE*EE);
  int rem = idx % (EE*EE);
  int n = rem / EE, k = rem % EE;
  const float* src = (w==0)?Wq:(w==1)?Wk:(w==2)?Wv:Wo;
  Wt[(size_t)w*EE*EE + (size_t)n*EE + k] = (_Float16)src[(size_t)k*EE + n];
}

// ---- 2. ref3d = sigmoid(query_table @ ref_W + ref_b); identical across batch ----
__global__ void k_ref3d(const float* qt, const float* refW, const float* refb,
                        float* out) {
  int idx = blockIdx.x*blockDim.x + threadIdx.x;
  if (idx >= QQ*3) return;
  int q = idx/3, j = idx%3;
  float acc = refb[j];
  for (int e = 0; e < EE; ++e) acc += qt[q*EE+e]*refW[e*3+j];
  float s = 1.f/(1.f + __expf(-acc));
  out[BB*QQ*10 + q*3 + j]        = s;   // b=0
  out[BB*QQ*10 + QQ*3 + q*3 + j] = s;   // b=1 (broadcast query table)
}

// ---- 3. lidar2img projection -> image-space sample coords + validity mask ----
__global__ void k_project(const float* l2i, const float* out, float* gbuf) {
  int idx = blockIdx.x*blockDim.x + threadIdx.x;
  if (idx >= BB*NN*QQ) return;
  int b = idx/(NN*QQ), n = (idx/QQ)%NN, q = idx%QQ;
  const float* r3 = out + BB*QQ*10 + (size_t)(b*QQ + q)*3;
  float X = r3[0], Y = r3[1], Z = r3[2];
  const float* L = l2i + (size_t)(b*NN+n)*16;
  float px = L[0]*X + L[1]*Y + L[2]*Z  + L[3];
  float py = L[4]*X + L[5]*Y + L[6]*Z  + L[7];
  float pz = L[8]*X + L[9]*Y + L[10]*Z + L[11];
  float az = fabsf(pz) + 1e-5f;
  float x2 = px/az, y2 = py/az;
  float gx = x2/(float)(IMW-1)*2.f - 1.f;
  float gy = y2/(float)(IMH-1)*2.f - 1.f;
  bool m = (pz > 1e-5f) && (gx > -1.f) && (gx < 1.f) && (gy > -1.f) && (gy < 1.f);
  gbuf[(size_t)idx*3+0] = ((gx+1.f)*(float)IMW - 1.f)*0.5f;
  gbuf[(size_t)idx*3+1] = ((gy+1.f)*(float)IMH - 1.f)*0.5f;
  gbuf[(size_t)idx*3+2] = m ? 1.f : 0.f;
}

// ---- 4. bilinear grid-sample + masked mean over cameras + residual -> x_f16 ----
__global__ void k_sample(const float* feats, const float* qt, const float* gbuf,
                         _Float16* xf) {
  int q = blockIdx.x, b = blockIdx.y, c = threadIdx.x;
  float acc = 0.f, cnt = 0.f;
  for (int n = 0; n < NN; ++n) {
    size_t gi = ((size_t)(b*NN+n)*QQ + q)*3;
    float m = gbuf[gi+2];
    cnt += m;
    if (m > 0.f) {
      float x = gbuf[gi+0], y = gbuf[gi+1];
      float x0f = floorf(x), y0f = floorf(y);
      float wx1 = x-x0f, wx0 = 1.f-wx1, wy1 = y-y0f, wy0 = 1.f-wy1;
      int x0 = (int)x0f, y0 = (int)y0f;
      const float* fb = feats + ((size_t)(b*NN+n)*CC + c)*(size_t)(IMH*IMW);
      auto g = [&](int xi, int yi) -> float {
        bool inb = (xi >= 0) && (xi < IMW) && (yi >= 0) && (yi < IMH);
        int xc = xi < 0 ? 0 : (xi > IMW-1 ? IMW-1 : xi);
        int yc = yi < 0 ? 0 : (yi > IMH-1 ? IMH-1 : yi);
        return inb ? fb[(size_t)yc*IMW + xc] : 0.f;
      };
      acc += g(x0,   y0  )*wx0*wy0 + g(x0+1, y0  )*wx1*wy0
           + g(x0,   y0+1)*wx0*wy1 + g(x0+1, y0+1)*wx1*wy1;
    }
  }
  if (cnt < 1.f) cnt = 1.f;
  float xv = qt[(size_t)q*EE + c] + acc/cnt;
  xf[((size_t)(b*QQ + q))*EE + c] = (_Float16)xv;
}

// ---- 5. QKV projection: one wave per 16x16 tile, WMMA f16 -> f32 accum ----
// q,k written as (B,NH,Q,hd); v written transposed as (B,NH,hd,VT_LD).
__global__ void k_qkv(const _Float16* xf, const _Float16* Wt,
                      const float* bq, const float* bk, const float* bv,
                      _Float16* qf, _Float16* kf, _Float16* vT) {
  const int lane = threadIdx.x;
  const int tm = blockIdx.x, tn = blockIdx.y, which = blockIdx.z;
  const _Float16* W = Wt + (size_t)which*EE*EE;
  const float* bias = (which==0)?bq:(which==1)?bk:bv;
  int m = tm*16 + (lane & 15);
  int arow = m < RTOT ? m : RTOT-1;
  int brow = tn*16 + (lane & 15);
  v8f c = {};
#pragma unroll
  for (int k0 = 0; k0 < EE; k0 += 32) {
    v16h a   = frag_load(xf, arow, EE, k0, lane);
    v16h bfr = frag_load(W,  brow, EE, k0, lane);
    c = __builtin_amdgcn_wmma_f32_16x16x32_f16(false, a, false, bfr,
                                               (short)0, c, false, false);
  }
  const int n = tn*16 + (lane & 15);
  const float bvv = bias[n];
  const int h = lane >> 4;
  const int head = n >> 5, d = n & 31;
#pragma unroll
  for (int v = 0; v < 8; ++v) {
    int r = tm*16 + v + 8*h;           // C layout: row = v + 8h, col = lane&15
    if (r < RTOT) {
      float val = c[v] + bvv;
      int b_ = r / QQ, qi = r % QQ;
      size_t bh = (size_t)(b_*NHH + head);
      if (which == 0)      qf[(bh*QQ + qi)*HDD + d]   = (_Float16)val;
      else if (which == 1) kf[(bh*QQ + qi)*HDD + d]   = (_Float16)val;
      else                 vT[(bh*HDD + d)*VT_LD + qi] = (_Float16)val;
    }
  }
}

// ---- 6. Flash attention: one wave per (16-query tile, b*head), online softmax ----
__global__ void k_flash(const _Float16* qf, const _Float16* kf,
                        const _Float16* vT, _Float16* of_) {
  __shared__ _Float16 lds[16*40];       // 16x32 P tile, stride 40 (16B aligned)
  const int lane = threadIdx.x;
  const int qt_ = blockIdx.x;           // 57 query tiles
  const int bh  = blockIdx.y;           // B*NH = 16
  const int h = lane >> 4, l = lane & 15;
  const _Float16* qb = qf + (size_t)bh*QQ*HDD;
  const _Float16* kb = kf + (size_t)bh*QQ*HDD;
  const _Float16* vb = vT + (size_t)bh*HDD*VT_LD;
  int qr = qt_*16 + l; if (qr > QQ-1) qr = QQ-1;
  v16h aq = frag_load(qb, qr, HDD, 0, lane);   // 16 queries x 32 dims
  float mrun[8], lrun[8];
  v8f o0 = {}, o1 = {};
#pragma unroll
  for (int v = 0; v < 8; ++v) { mrun[v] = -1e30f; lrun[v] = 0.f; }
  const float rs = 0.17677669529663687f;       // 1/sqrt(32)
  const int NKT = (QQ + 31)/32;                // 29 key tiles of 32
  for (int kt = 0; kt < NKT; ++kt) {
    int kbase = kt*32;
    int key0 = kbase + l;       bool inv0 = key0 >= QQ; int kr0 = inv0 ? QQ-1 : key0;
    int key1 = kbase + 16 + l;  bool inv1 = key1 >= QQ; int kr1 = inv1 ? QQ-1 : key1;
    v16h b0 = frag_load(kb, kr0, HDD, 0, lane);
    v16h b1 = frag_load(kb, kr1, HDD, 0, lane);
    v8f z = {};
    v8f s0 = __builtin_amdgcn_wmma_f32_16x16x32_f16(false, aq, false, b0,
                                                    (short)0, z, false, false);
    v8f s1 = __builtin_amdgcn_wmma_f32_16x16x32_f16(false, aq, false, b1,
                                                    (short)0, z, false, false);
    float p0[8], p1[8];
#pragma unroll
    for (int v = 0; v < 8; ++v) {
      float a0 = inv0 ? -1e30f : s0[v]*rs;
      float a1 = inv1 ? -1e30f : s1[v]*rs;
      float t = fmaxf(a0, a1);               // row max across the 16-lane half
      t = fmaxf(t, __shfl_xor(t, 1, 32));
      t = fmaxf(t, __shfl_xor(t, 2, 32));
      t = fmaxf(t, __shfl_xor(t, 4, 32));
      t = fmaxf(t, __shfl_xor(t, 8, 32));
      float mnew  = fmaxf(mrun[v], t);
      float alpha = __expf(mrun[v] - mnew);
      float e0 = __expf(a0 - mnew);
      float e1 = __expf(a1 - mnew);
      float rsum = e0 + e1;
      rsum += __shfl_xor(rsum, 1, 32);
      rsum += __shfl_xor(rsum, 2, 32);
      rsum += __shfl_xor(rsum, 4, 32);
      rsum += __shfl_xor(rsum, 8, 32);
      mrun[v] = mnew;
      lrun[v] = lrun[v]*alpha + rsum;
      o0[v] *= alpha; o1[v] *= alpha;
      p0[v] = e0; p1[v] = e1;
    }
    // C-layout -> A-layout transpose of P through LDS
#pragma unroll
    for (int v = 0; v < 8; ++v) {
      int row = v + 8*h;
      lds[row*40 + l]      = (_Float16)p0[v];
      lds[row*40 + 16 + l] = (_Float16)p1[v];
    }
    __syncthreads();
    v16h ap = frag_load(lds, l, 40, 0, lane);   // 16 queries x 32 keys
    __syncthreads();
    v16h bv0 = frag_load(vb, l,      VT_LD, kbase, lane);  // d = 0..15
    v16h bv1 = frag_load(vb, 16 + l, VT_LD, kbase, lane);  // d = 16..31
    o0 = __builtin_amdgcn_wmma_f32_16x16x32_f16(false, ap, false, bv0,
                                                (short)0, o0, false, false);
    o1 = __builtin_amdgcn_wmma_f32_16x16x32_f16(false, ap, false, bv1,
                                                (short)0, o1, false, false);
  }
  const int b_ = bh >> 3, head = bh & 7;
#pragma unroll
  for (int v = 0; v < 8; ++v) {
    int qrow = qt_*16 + v + 8*h;
    if (qrow < QQ) {
      float inv = 1.f / lrun[v];
      size_t base = ((size_t)(b_*QQ + qrow))*EE + head*HDD;
      of_[base + l]      = (_Float16)(o0[v]*inv);
      of_[base + 16 + l] = (_Float16)(o1[v]*inv);
    }
  }
}

// ---- 7. Output projection: x = o @ Wo + bo (f32 out for LN precision) ----
__global__ void k_oproj(const _Float16* of_, const _Float16* WtO,
                        const float* bo, float* xo) {
  const int lane = threadIdx.x;
  const int tm = blockIdx.x, tn = blockIdx.y;
  int m = tm*16 + (lane & 15);
  int arow = m < RTOT ? m : RTOT-1;
  int brow = tn*16 + (lane & 15);
  v8f c = {};
#pragma unroll
  for (int k0 = 0; k0 < EE; k0 += 32) {
    v16h a   = frag_load(of_, arow, EE, k0, lane);
    v16h bfr = frag_load(WtO, brow, EE, k0, lane);
    c = __builtin_amdgcn_wmma_f32_16x16x32_f16(false, a, false, bfr,
                                               (short)0, c, false, false);
  }
  int n = tn*16 + (lane & 15);
  float bb_ = bo[n];
  int h = lane >> 4;
#pragma unroll
  for (int v = 0; v < 8; ++v) {
    int r = tm*16 + v + 8*h;
    if (r < RTOT) xo[(size_t)r*EE + n] = c[v] + bb_;
  }
}

// ---- 8. LayerNorm + box head -> out[0:18000] ----
__global__ void k_ln_box(const float* xo, const float* g, const float* b,
                         const float* boxW, const float* boxb, float* out) {
  __shared__ float sb[256];
  int r = blockIdx.x, c = threadIdx.x;
  float x = xo[(size_t)r*EE + c];
  sb[c] = x; __syncthreads();
  for (int s = 128; s > 0; s >>= 1) { if (c < s) sb[c] += sb[c+s]; __syncthreads(); }
  float mu = sb[0]/(float)EE; __syncthreads();
  float d = x - mu;
  sb[c] = d*d; __syncthreads();
  for (int s = 128; s > 0; s >>= 1) { if (c < s) sb[c] += sb[c+s]; __syncthreads(); }
  float var = sb[0]/(float)EE; __syncthreads();
  float nv = d*rsqrtf(var + 1e-5f)*g[c] + b[c];
  for (int j = 0; j < 10; ++j) {
    sb[c] = nv*boxW[(size_t)c*10 + j]; __syncthreads();
    for (int s = 128; s > 0; s >>= 1) { if (c < s) sb[c] += sb[c+s]; __syncthreads(); }
    if (c == 0) out[(size_t)r*10 + j] = sb[0] + boxb[j];
    __syncthreads();
  }
}

extern "C" void kernel_launch(void* const* d_in, const int* in_sizes, int n_in,
                              void* d_out, int out_size, void* d_ws, size_t ws_size,
                              hipStream_t stream) {
  (void)in_sizes; (void)n_in; (void)out_size; (void)ws_size;
  const float* feats = (const float*)d_in[0];
  const float* l2i   = (const float*)d_in[1];
  const float* qt    = (const float*)d_in[2];
  const float* refW  = (const float*)d_in[3];
  const float* refb  = (const float*)d_in[4];
  const float* Wq    = (const float*)d_in[5];
  const float* bq    = (const float*)d_in[6];
  const float* Wk    = (const float*)d_in[7];
  const float* bk    = (const float*)d_in[8];
  const float* Wv    = (const float*)d_in[9];
  const float* bv    = (const float*)d_in[10];
  const float* Wo    = (const float*)d_in[11];
  const float* bo    = (const float*)d_in[12];
  const float* lng   = (const float*)d_in[13];
  const float* lnb   = (const float*)d_in[14];
  const float* boxW  = (const float*)d_in[15];
  const float* boxb  = (const float*)d_in[16];
  float* out = (float*)d_out;

  char* ws = (char*)d_ws;
  size_t off = 0;
  auto take = [&](size_t bytes) -> char* {
    char* p = ws + off;
    off += (bytes + 255) & ~(size_t)255;
    return p;
  };
  _Float16* Wt  = (_Float16*)take((size_t)4*EE*EE*sizeof(_Float16));      // 512 KB
  float*   gbuf = (float*)   take((size_t)BB*NN*QQ*3*sizeof(float));      // 127 KB
  _Float16* xf  = (_Float16*)take((size_t)RTOT*EE*sizeof(_Float16));      // 900 KB
  _Float16* qf  = (_Float16*)take((size_t)RTOT*EE*sizeof(_Float16));
  _Float16* kf  = (_Float16*)take((size_t)RTOT*EE*sizeof(_Float16));
  _Float16* vT  = (_Float16*)take(((size_t)BB*NHH*HDD*VT_LD + 64)*sizeof(_Float16));
  _Float16* of_ = (_Float16*)take((size_t)RTOT*EE*sizeof(_Float16));
  float*    xo  = (float*)   take((size_t)RTOT*EE*sizeof(float));         // 1.8 MB
  // total workspace ~7.2 MB

  k_prep_weights<<<dim3((4*EE*EE + 255)/256), 256, 0, stream>>>(Wq, Wk, Wv, Wo, Wt);
  k_ref3d<<<dim3((QQ*3 + 255)/256), 256, 0, stream>>>(qt, refW, refb, out);
  k_project<<<dim3((BB*NN*QQ + 255)/256), 256, 0, stream>>>(l2i, out, gbuf);
  k_sample<<<dim3(QQ, BB), 256, 0, stream>>>(feats, qt, gbuf, xf);
  k_qkv<<<dim3((RTOT + 15)/16, EE/16, 3), 32, 0, stream>>>(xf, Wt, bq, bk, bv,
                                                           qf, kf, vT);
  k_flash<<<dim3((QQ + 15)/16, BB*NHH), 32, 0, stream>>>(qf, kf, vT, of_);
  k_oproj<<<dim3((RTOT + 15)/16, EE/16), 32, 0, stream>>>(of_, Wt + (size_t)3*EE*EE,
                                                          bo, xo);
  k_ln_box<<<dim3(RTOT), 256, 0, stream>>>(xo, lng, lnb, boxW, boxb, out);
}
